// MultiHeadAttention_17987323036497
// MI455X (gfx1250) — compile-verified, matrix operand order checked by reading, not run on hardware
//
#include <hip/hip_runtime.h>
#include <stdint.h>

typedef __attribute__((ext_vector_type(16))) _Float16 v16h;
typedef __attribute__((ext_vector_type(8)))  _Float16 v8h;
typedef __attribute__((ext_vector_type(4)))  _Float16 v4h;
typedef __attribute__((ext_vector_type(8)))  float    v8f;
typedef __attribute__((ext_vector_type(4)))  float    v4f;

constexpr int   BATCH  = 2;
constexpr int   SEQ    = 2048;
constexpr int   HTOT   = 1024;
constexpr int   NHEADS = 16;
constexpr int   HDIM   = 64;
constexpr int   MTOT   = BATCH * SEQ;   // 4096 rows for the projection GEMMs
constexpr float NEGV   = -1000000.0f;

// ---------------------------------------------------------------------------
// WMMA helper: D = A(16x32 f16) * B(32x16 f16) + C(16x16 f32)
// ---------------------------------------------------------------------------
__device__ __forceinline__ v8f wmma16(v16h a, v16h b, v8f c) {
  return __builtin_amdgcn_wmma_f32_16x16x32_f16(
      /*neg_a=*/false, a, /*neg_b=*/false, b,
      /*c_mod=*/(short)0, c, /*reuse_a=*/false, /*reuse_b=*/false);
}

// CDNA5 async copy: global -> LDS, 16B per lane, tracked by ASYNCcnt.
// (inline asm per CDNA5 ISA; low 32 bits of a generic
//  LDS pointer are the LDS byte address per the aperture rules in 00_overview)
__device__ __forceinline__ void async_ld16(void* lds_dst, const void* gsrc) {
  uint32_t l = (uint32_t)(uintptr_t)lds_dst;
  asm volatile("global_load_async_to_lds_b128 %0, %1, off"
               :
               : "v"(l), "v"(gsrc)
               : "memory");
}
__device__ __forceinline__ void wait_async0() {
  asm volatile("s_wait_asynccnt 0x0" ::: "memory");
}

// Load one 16x32 f16 fragment (A-layout; B fragments are the mirrored layout,
// i.e. "row" = N column, which is why all B operands are stored K-major).
// lanes 0-15 -> row = lane, halves = K {0..7,16..23};
// lanes 16-31 -> row = lane-16, halves = K {8..15,24..31}.
__device__ __forceinline__ v16h ld_frag(const _Float16* __restrict__ p0, int stride) {
  int lane = threadIdx.x & 31;
  const _Float16* p = p0 + (lane & 15) * stride + ((lane >> 4) << 3);
  v8h lo = *(const v8h*)(p);
  v8h hi = *(const v8h*)(p + 16);
  v16h o;
#pragma unroll
  for (int i = 0; i < 8; ++i) { o[i] = lo[i]; o[i + 8] = hi[i]; }
  return o;
}

// ---------------------------------------------------------------------------
// Kernel 1: weight transpose + fp32->f16 convert.  Wt[n][k] = (f16) W[k][n]
// ---------------------------------------------------------------------------
__global__ __launch_bounds__(256) void transpose_w(const float* __restrict__ W,
                                                   _Float16* __restrict__ Wt) {
  __shared__ float tile[32][33];
  int bx = blockIdx.x * 32;        // n
  int by = blockIdx.y * 32;        // k
  int tx = threadIdx.x & 31;
  int ty = threadIdx.x >> 5;       // 0..7
#pragma unroll
  for (int i = ty; i < 32; i += 8)
    tile[i][tx] = W[(size_t)(by + i) * HTOT + bx + tx];
  __syncthreads();
#pragma unroll
  for (int i = ty; i < 32; i += 8)
    Wt[(size_t)(bx + i) * HTOT + by + tx] = (_Float16)tile[tx][i];
}

// ---------------------------------------------------------------------------
// Kernel 2: WMMA GEMM  C[M,N] = A[M,K] * Bt[N,K]^T + bias
//   A_F16=0: A fp32 (converted while staging to LDS); A_F16=1: A f16 (async).
//   MODE 0: f16 out, head-split  [B,NH,S,HD]   (Q,K projections)
//   MODE 1: f16 out, transposed  [B,NH,HD,S]   (V projection)
//   MODE 2: f32 out, row-major   [M,N]         (final output projection)
// Block: 256 threads = 8 waves; block tile 128x64; wave tile 32x32 (2x2 WMMA).
// Double-buffered LDS, async global->LDS prefetch of the next K tile.
// ---------------------------------------------------------------------------
template <int A_F16, int MODE>
__global__ __launch_bounds__(256) void gemm_wmma(
    const void* __restrict__ Aptr, const _Float16* __restrict__ Bt,
    const float* __restrict__ bias, void* __restrict__ Cptr,
    int M, int N, int K) {
  __shared__ __align__(16) _Float16 As[2][128 * 40];  // 128x32, pad to 40 halves
  __shared__ __align__(16) _Float16 Bs[2][64 * 40];   // 64x32

  int t    = threadIdx.x;
  int lane = t & 31;
  int wave = t >> 5;
  int bm   = blockIdx.x * 128;
  int bn   = blockIdx.y * 64;
  int wm   = (wave >> 1) * 32;
  int wn   = (wave & 1) * 32;

  auto stage = [&](int kb, int buf) {
    if (A_F16) {
      const _Float16* A = (const _Float16*)Aptr;
#pragma unroll
      for (int i = 0; i < 2; ++i) {
        int c = t + i * 256;
        int row = c >> 2, cp = c & 3;
        async_ld16(&As[buf][row * 40 + cp * 8],
                   A + (size_t)(bm + row) * K + kb + cp * 8);
      }
    } else {
      const float* A = (const float*)Aptr;
#pragma unroll
      for (int i = 0; i < 4; ++i) {
        int c = t + i * 256;
        int row = c >> 3, cp = c & 7;
        v4f d = *(const v4f*)(A + (size_t)(bm + row) * K + kb + cp * 4);
        v4h hd;
#pragma unroll
        for (int e = 0; e < 4; ++e) hd[e] = (_Float16)d[e];
        *(v4h*)(&As[buf][row * 40 + cp * 4]) = hd;
      }
    }
    {
      int row = t >> 2, cp = t & 3;
      async_ld16(&Bs[buf][row * 40 + cp * 8],
                 Bt + (size_t)(bn + row) * K + kb + cp * 8);
    }
  };

  v8f acc00 = {}, acc01 = {}, acc10 = {}, acc11 = {};

  stage(0, 0);
  wait_async0();
  __syncthreads();

  for (int kb = 0; kb < K; kb += 32) {
    int cur = (kb >> 5) & 1;
    if (kb + 32 < K) stage(kb + 32, cur ^ 1);   // prefetch next tile

    v16h a0 = ld_frag(&As[cur][(wm + 0)  * 40], 40);
    v16h a1 = ld_frag(&As[cur][(wm + 16) * 40], 40);
    v16h b0 = ld_frag(&Bs[cur][(wn + 0)  * 40], 40);
    v16h b1 = ld_frag(&Bs[cur][(wn + 16) * 40], 40);
    acc00 = wmma16(a0, b0, acc00);
    acc01 = wmma16(a0, b1, acc01);
    acc10 = wmma16(a1, b0, acc10);
    acc11 = wmma16(a1, b1, acc11);

    wait_async0();
    __syncthreads();
  }

  // ---- epilogue: bias + layout-specific store ----
  int ln = lane & 15;
  int lm = (lane >> 4) * 8;
  v8f accs[2][2] = {{acc00, acc01}, {acc10, acc11}};
#pragma unroll
  for (int i = 0; i < 2; ++i) {
#pragma unroll
    for (int j = 0; j < 2; ++j) {
      int n = bn + wn + j * 16 + ln;
      float bv = bias[n];
      v8f c = accs[i][j];
#pragma unroll
      for (int r = 0; r < 8; ++r) {
        int m = bm + wm + i * 16 + lm + r;
        float val = c[r] + bv;
        if (MODE == 2) {
          ((float*)Cptr)[(size_t)m * N + n] = val;
        } else {
          int bi = m >> 11, s = m & (SEQ - 1);
          int hi = n >> 6,  d = n & (HDIM - 1);
          size_t idx = (MODE == 0)
              ? ((((size_t)bi * NHEADS + hi) * SEQ) + s) * HDIM + d
              : ((((size_t)bi * NHEADS + hi) * HDIM) + d) * SEQ + s;
          ((_Float16*)Cptr)[idx] = (_Float16)val;
        }
      }
    }
  }
}

// ---------------------------------------------------------------------------
// Kernel 3: flash-style masked attention with online softmax.
//   Qh, Kh: f16 [B,NH,SEQ,HDIM] ; Vt: f16 [B,NH,HDIM,SEQ] ; mask int32 [SEQ,SEQ]
//   O: f16 [B,SEQ,HTOT] (head-interleaved, ready for the output projection)
// Block: 256 threads = 8 waves; each wave owns 16 query rows; KV tile = 32.
// K/V tiles staged to double-buffered LDS with async copies (shared by the
// whole block instead of 8x redundant global fetches).
// ---------------------------------------------------------------------------
__global__ __launch_bounds__(256) void attn_wmma(
    const _Float16* __restrict__ Qh, const _Float16* __restrict__ Kh,
    const _Float16* __restrict__ Vt, const int* __restrict__ mask,
    _Float16* __restrict__ O) {
  __shared__ __align__(16) _Float16 Ks[2][32 * 72];   // 32 kv rows x 64 hd
  __shared__ __align__(16) _Float16 Vs[2][64 * 40];   // 64 d rows x 32 kv
  __shared__ __align__(16) _Float16 Pl[8 * 16 * 40];  // per-wave 16x32 P tile

  int t    = threadIdx.x;
  int lane = t & 31;
  int wave = t >> 5;
  int bh   = blockIdx.y;             // b*NH + h
  int bi   = bh >> 4;
  int hi   = bh & (NHEADS - 1);
  int qw   = blockIdx.x * 128 + wave * 16;  // this wave's first query row
  int ln   = lane & 15;
  int lm   = (lane >> 4) * 8;

  const _Float16* Qp = Qh + (size_t)bh * SEQ * HDIM;
  const _Float16* Kp = Kh + (size_t)bh * SEQ * HDIM;
  const _Float16* Vp = Vt + (size_t)bh * HDIM * SEQ;
  _Float16* Pw = &Pl[wave * 16 * 40];

  auto stage = [&](int kv, int buf) {
    {  // K tile: 32 rows x 64 halves = 256 x 16B chunks
      int row = t >> 3, cp = t & 7;
      async_ld16(&Ks[buf][row * 72 + cp * 8],
                 Kp + (size_t)(kv + row) * HDIM + cp * 8);
    }
    {  // V tile: 64 d-rows x 32 halves = 256 x 16B chunks
      int row = t >> 2, cp = t & 3;
      async_ld16(&Vs[buf][row * 40 + cp * 8],
                 Vp + (size_t)row * SEQ + kv + cp * 8);
    }
  };

  // Q fragments for this wave's 16 rows, K = 0..63 (register resident).
  v16h qf0 = ld_frag(Qp + (size_t)qw * HDIM + 0,  HDIM);
  v16h qf1 = ld_frag(Qp + (size_t)qw * HDIM + 32, HDIM);

  v8f o0 = {}, o1 = {}, o2 = {}, o3 = {};
  float mrow[8], lrow[8];
#pragma unroll
  for (int r = 0; r < 8; ++r) { mrow[r] = -3.0e38f; lrow[r] = 0.0f; }

  const float scale = 0.125f;  // 1/sqrt(64)

  stage(0, 0);
  wait_async0();
  __syncthreads();

  for (int kv = 0; kv < SEQ; kv += 32) {
    int cur = (kv >> 5) & 1;
    if (kv + 32 < SEQ) stage(kv + 32, cur ^ 1);   // prefetch next KV tile

    // ---- S = Q * K^T for a 16x32 tile (4 WMMAs, K tile from LDS) ----
    v16h k0a = ld_frag(&Ks[cur][0 * 72 + 0],  72);
    v16h k0b = ld_frag(&Ks[cur][0 * 72 + 32], 72);
    v16h k1a = ld_frag(&Ks[cur][16 * 72 + 0],  72);
    v16h k1b = ld_frag(&Ks[cur][16 * 72 + 32], 72);
    v8f s0 = {}, s1 = {};
    s0 = wmma16(qf0, k0a, s0);
    s0 = wmma16(qf1, k0b, s0);
    s1 = wmma16(qf0, k1a, s1);
    s1 = wmma16(qf1, k1b, s1);

    // ---- mask + online softmax update ----
    float al[8];
#pragma unroll
    for (int r = 0; r < 8; ++r) {
      int qr = qw + lm + r;
      int m0 = mask[(size_t)qr * SEQ + kv + ln];
      int m1 = mask[(size_t)qr * SEQ + kv + 16 + ln];
      float v0 = m0 ? s0[r] * scale : NEGV;
      float v1 = m1 ? s1[r] * scale : NEGV;
      float tm = fmaxf(v0, v1);
#pragma unroll
      for (int off = 1; off < 16; off <<= 1)
        tm = fmaxf(tm, __shfl_xor(tm, off, 16));
      float mn = fmaxf(mrow[r], tm);
      al[r]    = __expf(mrow[r] - mn);
      mrow[r]  = mn;
      float p0 = __expf(v0 - mn);
      float p1 = __expf(v1 - mn);
      float ps = p0 + p1;
#pragma unroll
      for (int off = 1; off < 16; off <<= 1)
        ps += __shfl_xor(ps, off, 16);
      lrow[r] = lrow[r] * al[r] + ps;
      // stash P as f16 in this wave's LDS tile (accumulator -> A-frag relayout)
      Pw[(lm + r) * 40 + ln]      = (_Float16)p0;
      Pw[(lm + r) * 40 + 16 + ln] = (_Float16)p1;
    }
#pragma unroll
    for (int r = 0; r < 8; ++r) {
      o0[r] *= al[r]; o1[r] *= al[r]; o2[r] *= al[r]; o3[r] *= al[r];
    }

    wait_async0();       // prefetched tile landed
    __syncthreads();     // P visible; stage(buf^1) complete for next iter

    // ---- O += P(16x32) * V(32x64)  (4 WMMAs; V tile from LDS) ----
    v16h pf  = ld_frag(Pw, 40);
    v16h vb0 = ld_frag(&Vs[cur][0  * 40], 40);
    v16h vb1 = ld_frag(&Vs[cur][16 * 40], 40);
    v16h vb2 = ld_frag(&Vs[cur][32 * 40], 40);
    v16h vb3 = ld_frag(&Vs[cur][48 * 40], 40);
    o0 = wmma16(pf, vb0, o0);
    o1 = wmma16(pf, vb1, o1);
    o2 = wmma16(pf, vb2, o2);
    o3 = wmma16(pf, vb3, o3);

    __syncthreads();     // V reads done before next iteration's stage overwrites
  }

  // ---- normalize and store head-interleaved f16 ----
#pragma unroll
  for (int r = 0; r < 8; ++r) {
    float inv = 1.0f / lrow[r];
    size_t base = ((size_t)bi * SEQ + qw + lm + r) * HTOT + hi * HDIM;
    O[base + 0  + ln] = (_Float16)(o0[r] * inv);
    O[base + 16 + ln] = (_Float16)(o1[r] * inv);
    O[base + 32 + ln] = (_Float16)(o2[r] * inv);
    O[base + 48 + ln] = (_Float16)(o3[r] * inv);
  }
}

// ---------------------------------------------------------------------------
// Host launch
// ---------------------------------------------------------------------------
extern "C" void kernel_launch(void* const* d_in, const int* in_sizes, int n_in,
                              void* d_out, int out_size, void* d_ws, size_t ws_size,
                              hipStream_t stream) {
  const float* q    = (const float*)d_in[0];
  const float* k    = (const float*)d_in[1];
  const float* v    = (const float*)d_in[2];
  const int*   mask = (const int*)  d_in[3];
  const float* wq   = (const float*)d_in[4];
  const float* bq   = (const float*)d_in[5];
  const float* wk   = (const float*)d_in[6];
  const float* bk   = (const float*)d_in[7];
  const float* wv   = (const float*)d_in[8];
  const float* bv   = (const float*)d_in[9];
  const float* wo   = (const float*)d_in[10];
  const float* bo   = (const float*)d_in[11];
  float*       out  = (float*)d_out;

  char* ws = (char*)d_ws;
  const size_t MiB = 1024 * 1024;
  _Float16* wqT = (_Float16*)(ws + 0 * MiB);   // 1024*1024 f16 = 2 MiB each
  _Float16* wkT = (_Float16*)(ws + 2 * MiB);
  _Float16* wvT = (_Float16*)(ws + 4 * MiB);
  _Float16* woT = (_Float16*)(ws + 6 * MiB);
  _Float16* Qh  = (_Float16*)(ws + 8 * MiB);   // [B,NH,S,HD] f16 = 8 MiB each
  _Float16* Kh  = (_Float16*)(ws + 16 * MiB);
  _Float16* Vt  = (_Float16*)(ws + 24 * MiB);  // [B,NH,HD,S]
  _Float16* O16 = (_Float16*)(ws + 32 * MiB);  // [B,S,H]     (total 40 MiB)

  dim3 tb(256);

  // 1) pre-transpose + convert weights to f16 (K-major for WMMA B fragments)
  dim3 tg(HTOT / 32, HTOT / 32);
  transpose_w<<<tg, tb, 0, stream>>>(wq, wqT);
  transpose_w<<<tg, tb, 0, stream>>>(wk, wkT);
  transpose_w<<<tg, tb, 0, stream>>>(wv, wvT);
  transpose_w<<<tg, tb, 0, stream>>>(wo, woT);

  // 2) QKV projections (fp32 A converted in-kernel)
  dim3 gg(MTOT / 128, HTOT / 64);
  gemm_wmma<0, 0><<<gg, tb, 0, stream>>>((const void*)q, wqT, bq, (void*)Qh,
                                         MTOT, HTOT, HTOT);
  gemm_wmma<0, 0><<<gg, tb, 0, stream>>>((const void*)k, wkT, bk, (void*)Kh,
                                         MTOT, HTOT, HTOT);
  gemm_wmma<0, 1><<<gg, tb, 0, stream>>>((const void*)v, wvT, bv, (void*)Vt,
                                         MTOT, HTOT, HTOT);

  // 3) masked attention with online softmax
  dim3 ag(SEQ / 128, BATCH * NHEADS);
  attn_wmma<<<ag, tb, 0, stream>>>(Qh, Kh, Vt, mask, O16);

  // 4) output projection -> fp32 d_out
  gemm_wmma<1, 2><<<gg, tb, 0, stream>>>((const void*)O16, woT, bo, (void*)out,
                                         MTOT, HTOT, HTOT);
}
